// Self_RNN_25950192402646
// MI455X (gfx1250) — matrix operand order repaired and checked by reading, not compiled, not run on hardware
//
#include <hip/hip_runtime.h>
#include <hip/hip_bf16.h>

// ---------------------------------------------------------------------------
// Self_RNN on gfx1250 (MI455X), wave32 + v_wmma_f32_16x16x32_bf16.
//
//   Wrqkv = Wrec @ [Wq|Wk|Wv]                (512 x 1536, folded once)
//   xQV   = x @ [Wq|Wv]                      (B*T x 1024, parallel precompute)
//   per step t:  P = h @ Wrqkv  -> [Qh|K0|Vh]  (64 x 1536)
//     s0 = K0.Qh, s1 = K0.xQ[t-1], s2 = K0.xQ[t]  (s1,Vprev = 0 at t==0)
//     a = softmax(s * 1/sqrt(512)); h' = a0*Vh + a1*xV[t-1] + a2*xV[t]
//
// h is carried in bf16 (it is only consumed as a WMMA A-operand); all
// accumulation, softmax and the final output are fp32.
// ---------------------------------------------------------------------------

typedef __attribute__((ext_vector_type(16))) __bf16 v16bf;
typedef __attribute__((ext_vector_type(8)))  __bf16 v8bf;
typedef __attribute__((ext_vector_type(4)))  __bf16 v4bf;
typedef __attribute__((ext_vector_type(8)))  float  v8f;
typedef __attribute__((ext_vector_type(4)))  float  v4f;
typedef __attribute__((ext_vector_type(4)))  int    v4i;

#define BATCH 64
#define TSTEPS 512
#define DIM 512
#define N3 1536          // Wrq|Wrk|Wrv columns
#define NQV 1024         // xQ|xV columns
#define NBLK 96          // persistent grid (must be co-resident)
#define INV_SQRT_U 0.04419417382415922f

#if __has_builtin(__builtin_amdgcn_global_load_async_to_lds_b128)
#define HAVE_ASYNC_LDS 1
#endif

// ---------------------------------------------------------------------------
// Fragment helpers (layouts per CDNA5 ISA 7.12.2, wave32)
// ---------------------------------------------------------------------------

// A fragment (16x32 bf16) from a bf16 row-major source: two 8-element runs
// K = kb + 8*half + [0,8) and K = kb + 16 + 8*half + [0,8).
__device__ __forceinline__ v16bf a_frag_bf16(const __bf16* __restrict__ arow,
                                             int kb, int half) {
  const int hB = half * 8;
  v8bf r0 = *(const v8bf*)(arow + kb + hB);
  v8bf r1 = *(const v8bf*)(arow + kb + hB + 16);
  v16bf a;
#pragma unroll
  for (int e = 0; e < 8; ++e) { a[e] = r0[e]; a[8 + e] = r1[e]; }
  return a;
}

// B fragment (32x16 bf16): lane L holds column n=n0+L%16, K = kb + 8*half + [0,8)
// and K = kb + 16 + 8*half + [0,8).
__device__ __forceinline__ v16bf b_frag(const __bf16* __restrict__ Bw,
                                        int kb, int lane, int ldb, int n0) {
  const int hB = (lane >> 4) * 8, n = n0 + (lane & 15);
  v16bf b;
#pragma unroll
  for (int e = 0; e < 8; ++e) {
    b[e] = Bw[(size_t)(kb + hB + e) * ldb + n];
    b[8 + e] = Bw[(size_t)(kb + hB + 16 + e) * ldb + n];
  }
  return b;
}

// ---------------------------------------------------------------------------
// Setup kernels
// ---------------------------------------------------------------------------

__global__ void init_kernel(__bf16* h0, __bf16* h1, unsigned* sync) {
  int i = blockIdx.x * blockDim.x + threadIdx.x;
  if (i < BATCH * DIM) { h0[i] = (__bf16)0.0f; h1[i] = (__bf16)0.0f; }
  if (i == 0) { sync[0] = 0u; sync[1] = 0u; }
}

__global__ void cvt_x_kernel(const float* __restrict__ x, __bf16* __restrict__ xbf,
                             int n4) {
  int i = blockIdx.x * blockDim.x + threadIdx.x;
  if (i >= n4) return;
  v4f f = *(const v4f*)(x + 4 * (size_t)i);
  v4bf o;
  o[0] = (__bf16)f[0]; o[1] = (__bf16)f[1]; o[2] = (__bf16)f[2]; o[3] = (__bf16)f[3];
  *(v4bf*)(xbf + 4 * (size_t)i) = o;
}

__global__ void pack_w_kernel(const float* __restrict__ Wq, const float* __restrict__ Wk,
                              const float* __restrict__ Wv, const float* __restrict__ Wr,
                              __bf16* __restrict__ Wqkv, __bf16* __restrict__ Wrec) {
  int i = blockIdx.x * blockDim.x + threadIdx.x;
  const int n1 = DIM * N3;
  if (i < n1) {
    int k = i / N3, c = i % N3;
    int sel = c >> 9, col = c & 511;
    float v = (sel == 0 ? Wq : (sel == 1 ? Wk : Wv))[k * DIM + col];
    Wqkv[i] = (__bf16)v;
  } else if (i < n1 + DIM * DIM) {
    int j = i - n1;
    Wrec[j] = (__bf16)Wr[j];
  }
}

// Wrqkv(bf16, 512x1536) = Wrec(bf16, 512x512) @ Wqkv(bf16, 512x1536)
__global__ void gemm_fold_kernel(const __bf16* __restrict__ A,
                                 const __bf16* __restrict__ Bw,
                                 __bf16* __restrict__ C) {
  const int lane = threadIdx.x & 31, wave = threadIdx.x >> 5;
  const int tile = blockIdx.x * 4 + wave;             // 3072 tiles
  const int rt = tile / 96, ct = tile % 96;
  const int half = lane >> 4, nl = lane & 15;
  const __bf16* arow = A + (size_t)(rt * 16 + nl) * DIM;
  v8f acc = {};
#pragma unroll 4
  for (int kk = 0; kk < 16; ++kk) {
    v16bf a = a_frag_bf16(arow, kk * 32, half);
    v16bf b = b_frag(Bw, kk * 32, lane, N3, ct * 16);
    acc = __builtin_amdgcn_wmma_f32_16x16x32_bf16(false, a, false, b,
                                                  (short)0, acc, false, false);
  }
  __bf16* crow = C + (size_t)(rt * 16 + half * 8) * N3 + ct * 16 + nl;
#pragma unroll
  for (int v = 0; v < 8; ++v) crow[(size_t)v * N3] = (__bf16)acc[v];
}

// xQV(fp32, 32768x1024) = xbf(32768x512) @ [Wq|Wv] taken from Wqkv(512x1536).
// Each wave computes TWO adjacent 16x16 output tiles, reusing the A fragment.
// Job j in [0,64): weight col tile cw = (j<32 ? j : j+32)  (skips the Wk block).
__global__ void gemm_xqv_kernel(const __bf16* __restrict__ A,
                                const __bf16* __restrict__ Bw,
                                float* __restrict__ C) {
  const int lane = threadIdx.x & 31, wave = threadIdx.x >> 5;
  const int tile = blockIdx.x * 4 + wave;             // 65536 wave-jobs
  const int rt = tile >> 5;                           // 0..2047 row tiles
  const int jj = tile & 31;                           // job pair
  const int j0 = jj * 2, j1 = j0 + 1;
  const int cw0 = (j0 < 32) ? j0 : j0 + 32;
  const int cw1 = (j1 < 32) ? j1 : j1 + 32;
  const int half = lane >> 4, nl = lane & 15;
  const __bf16* arow = A + (size_t)(rt * 16 + nl) * DIM;
  v8f acc0 = {}, acc1 = {};
#pragma unroll 4
  for (int kk = 0; kk < 16; ++kk) {
    if (kk < 15)  // -> global_prefetch_b8: hide A latency for next k-slice
      __builtin_prefetch(arow + (kk + 1) * 32 + half * 8, 0, 0);
    v16bf a  = a_frag_bf16(arow, kk * 32, half);
    v16bf b0 = b_frag(Bw, kk * 32, lane, N3, cw0 * 16);
    v16bf b1 = b_frag(Bw, kk * 32, lane, N3, cw1 * 16);
    acc0 = __builtin_amdgcn_wmma_f32_16x16x32_bf16(false, a, false, b0,
                                                   (short)0, acc0, false, false);
    acc1 = __builtin_amdgcn_wmma_f32_16x16x32_bf16(false, a, false, b1,
                                                   (short)0, acc1, false, false);
  }
  float* crow = C + (size_t)(rt * 16 + half * 8) * NQV + j0 * 16 + nl;
#pragma unroll
  for (int v = 0; v < 8; ++v) {
    crow[(size_t)v * NQV]      = acc0[v];
    crow[(size_t)v * NQV + 16] = acc1[v];
  }
}

// ---------------------------------------------------------------------------
// Persistent recurrence kernel: 96 blocks x 128 threads (4 wave32).
// ---------------------------------------------------------------------------

__device__ __forceinline__ void grid_sync(unsigned* cnt, unsigned* gen) {
  __threadfence();          // release our global writes (device scope)
  __syncthreads();
  if (threadIdx.x == 0) {
    unsigned g = __hip_atomic_load(gen, __ATOMIC_ACQUIRE, __HIP_MEMORY_SCOPE_AGENT);
    unsigned a = __hip_atomic_fetch_add(cnt, 1u, __ATOMIC_ACQ_REL, __HIP_MEMORY_SCOPE_AGENT);
    if (a == NBLK - 1) {
      __hip_atomic_store(cnt, 0u, __ATOMIC_RELAXED, __HIP_MEMORY_SCOPE_AGENT);
      __hip_atomic_fetch_add(gen, 1u, __ATOMIC_ACQ_REL, __HIP_MEMORY_SCOPE_AGENT);
    } else {
      while (__hip_atomic_load(gen, __ATOMIC_ACQUIRE, __HIP_MEMORY_SCOPE_AGENT) == g)
        __builtin_amdgcn_s_sleep(1);
    }
  }
  __syncthreads();
}

__global__ void rnn_persistent_kernel(const float* __restrict__ xQV,
                                      const __bf16* __restrict__ Wrqkv,
                                      __bf16* h0buf, __bf16* h1buf,
                                      float* __restrict__ P,
                                      float* __restrict__ out,
                                      unsigned* sync) {
  const int g = blockIdx.x;
  const int tid = threadIdx.x;
  const int lane = tid & 31, wave = tid >> 5;
  const int half = lane >> 4, nl = lane & 15;
  __shared__ __bf16 Bs[DIM * 16];          // this block's 512x16 weight slice (16 KB)
  __shared__ float red0[128], red1[128], red2[128];
  __shared__ float attnS[3];

  // ---- Stage the constant weight slice for column tile g into LDS once. ----
  // B-fragment layout (lane = col n, row k = 16 contiguous cols) == row-major, so the
  // copy is a straight 32 B/row gather of columns [16g, 16g+16).
  {
    const __bf16* src0 = Wrqkv + g * 16;
    for (int c = tid; c < 2 * DIM; c += 128) {      // 16-byte chunks
      int r = c >> 1, e = (c & 1) << 3;
      const __bf16* gp = src0 + (size_t)r * N3 + e;
      __bf16* lp = Bs + r * 16 + e;
#ifdef HAVE_ASYNC_LDS
      __builtin_amdgcn_global_load_async_to_lds_b128(
          (__attribute__((address_space(1))) v4i*)gp,
          (__attribute__((address_space(3))) v4i*)lp, 0, 0);
#else
      *(v8bf*)lp = *(const v8bf*)gp;
#endif
    }
#ifdef HAVE_ASYNC_LDS
#if __has_builtin(__builtin_amdgcn_s_wait_asynccnt)
    __builtin_amdgcn_s_wait_asynccnt(0);
#else
    asm volatile("s_wait_asynccnt 0x0" ::: "memory");
#endif
#endif
    __syncthreads();
  }

  for (int t = 0; t < TSTEPS; ++t) {
    const __bf16* h  = (t & 1) ? h1buf : h0buf;
    __bf16*       hn = (t & 1) ? h0buf : h1buf;

    // ---- Phase A: P(64x1536) = h(64x512) @ Wrqkv(512x1536), B from LDS ----
    {
      const int rt = wave;         // row tile 0..3
      const __bf16* arow = h + (size_t)(rt * 16 + nl) * DIM;
      v8f acc = {};
#pragma unroll 4
      for (int kk = 0; kk < 16; ++kk) {
        v16bf a = a_frag_bf16(arow, kk * 32, half);
        v16bf b;
#pragma unroll
        for (int e = 0; e < 8; ++e) {
          b[e] = Bs[(kk * 32 + half * 8 + e) * 16 + nl];
          b[8 + e] = Bs[(kk * 32 + half * 8 + 16 + e) * 16 + nl];
        }
        acc = __builtin_amdgcn_wmma_f32_16x16x32_bf16(false, a, false, b,
                                                      (short)0, acc, false, false);
      }
      float* crow = P + (size_t)(rt * 16 + half * 8) * N3 + g * 16 + nl;
#pragma unroll
      for (int v = 0; v < 8; ++v) crow[(size_t)v * N3] = acc[v];
    }
    grid_sync(sync, sync + 1);

    // ---- Phase B: scores -> softmax -> combine (blocks 0..63, one batch each) ----
    if (g < BATCH) {
      const int b = g;
      const float* Qh  = P + (size_t)b * N3;
      const float* K0  = Qh + DIM;
      const float* Vh  = Qh + 2 * DIM;
      const float* xQc = xQV + (size_t)(b * TSTEPS + t) * NQV;
      const float* xVc = xQc + DIM;
      const float* xQp = xQc - NQV;   // valid only for t>0
      const float* xVp = xVc - NQV;

      float s0 = 0.f, s1 = 0.f, s2 = 0.f;
      for (int i = tid; i < DIM; i += 128) {
        float k = K0[i];
        s0 += k * Qh[i];
        s2 += k * xQc[i];
        if (t > 0) s1 += k * xQp[i];
      }
      red0[tid] = s0; red1[tid] = s1; red2[tid] = s2;
      __syncthreads();
      for (int s = 64; s > 0; s >>= 1) {
        if (tid < s) {
          red0[tid] += red0[tid + s];
          red1[tid] += red1[tid + s];
          red2[tid] += red2[tid + s];
        }
        __syncthreads();
      }
      if (tid == 0) {
        float z0 = red0[0] * INV_SQRT_U;
        float z1 = (t > 0) ? red1[0] * INV_SQRT_U : 0.0f;  // K0 . (0 @ Wq) = 0
        float z2 = red2[0] * INV_SQRT_U;
        float mx = fmaxf(z0, fmaxf(z1, z2));
        float e0 = __expf(z0 - mx), e1 = __expf(z1 - mx), e2 = __expf(z2 - mx);
        float inv = 1.0f / (e0 + e1 + e2);
        attnS[0] = e0 * inv; attnS[1] = e1 * inv; attnS[2] = e2 * inv;
      }
      __syncthreads();
      float a0 = attnS[0], a1 = attnS[1], a2 = attnS[2];
      for (int i = tid; i < DIM; i += 128) {
        float o = a0 * Vh[i] + a2 * xVc[i];
        if (t > 0) o += a1 * xVp[i];
        hn[(size_t)b * DIM + i] = (__bf16)o;            // carried state in bf16
        if (t == TSTEPS - 1) out[(size_t)b * DIM + i] = o;  // final output fp32
      }
    }
    grid_sync(sync, sync + 1);
  }
}

// ---------------------------------------------------------------------------
// Host side
// ---------------------------------------------------------------------------

extern "C" void kernel_launch(void* const* d_in, const int* in_sizes, int n_in,
                              void* d_out, int out_size, void* d_ws, size_t ws_size,
                              hipStream_t stream) {
  const float* x   = (const float*)d_in[0];
  const float* Wq  = (const float*)d_in[1];
  const float* Wk  = (const float*)d_in[2];
  const float* Wv  = (const float*)d_in[3];
  const float* Wr  = (const float*)d_in[4];
  float* out = (float*)d_out;

  // Workspace carve-up (~172 MB total).
  uint8_t* w = (uint8_t*)d_ws;
  unsigned* sync  = (unsigned*)w;                       // 256 B
  float* P        = (float*)(w + 256);                  // 64 x 1536
  float* xQV      = P + (size_t)BATCH * N3;             // 32768 x 1024
  __bf16* h0      = (__bf16*)(xQV + (size_t)BATCH * TSTEPS * NQV);
  __bf16* h1      = h0 + BATCH * DIM;
  __bf16* Wqkv    = h1 + BATCH * DIM;                   // 512 x 1536
  __bf16* WrecBf  = Wqkv + (size_t)DIM * N3;            // 512 x 512
  __bf16* Wrqkv   = WrecBf + (size_t)DIM * DIM;         // 512 x 1536
  __bf16* xbf     = Wrqkv + (size_t)DIM * N3;           // 32768 x 512

  // 1) zero h + sync
  init_kernel<<<(BATCH * DIM + 255) / 256, 256, 0, stream>>>(h0, h1, sync);
  // 2) x -> bf16 (vectorized x4)
  const int n4 = BATCH * TSTEPS * DIM / 4;
  cvt_x_kernel<<<(n4 + 255) / 256, 256, 0, stream>>>(x, xbf, n4);
  // 3) pack weights -> bf16
  const int npw = DIM * N3 + DIM * DIM;
  pack_w_kernel<<<(npw + 255) / 256, 256, 0, stream>>>(Wq, Wk, Wv, Wr, Wqkv, WrecBf);
  // 4) fold recurrent weights: Wrqkv = Wrec @ [Wq|Wk|Wv]  (3072 tiles / 4 waves)
  gemm_fold_kernel<<<768, 128, 0, stream>>>(WrecBf, Wqkv, Wrqkv);
  // 5) xQV = x @ [Wq|Wv]  (65536 wave-jobs, 2 tiles per wave)
  gemm_xqv_kernel<<<16384, 128, 0, stream>>>(xbf, Wqkv, xQV);
  // 6) persistent recurrence over 512 steps
  rnn_persistent_kernel<<<NBLK, 128, 0, stream>>>(xQV, Wrqkv, h0, h1, P, out, sync);
}
